// HierarchicalTransformerEncoder_36464272343618
// MI455X (gfx1250) — matrix operand order repaired and checked by reading, not compiled, not run on hardware
//
#include <hip/hip_runtime.h>
#include <hip/hip_bf16.h>

// ---------------------------------------------------------------------------
// SegFormer encoder block for MI455X (gfx1250, wave32, WMMA + TDM).
// All big GEMMs -> v_wmma_f32_16x16x32_f16 (f32 accumulate), weight slabs
// staged into LDS by the Tensor Data Mover (double-buffered, TENSORcnt-
// pipelined). Attention is fused flash-attention with 64-wide kv tiles.
// ---------------------------------------------------------------------------

typedef __attribute__((ext_vector_type(16))) _Float16     v16h;
typedef __attribute__((ext_vector_type(8)))  float        v8f;
typedef __attribute__((ext_vector_type(4)))  unsigned int v4u;
typedef __attribute__((ext_vector_type(8)))  int          v8i;
typedef __attribute__((ext_vector_type(4)))  int          v4i;

#define WMMA_F16(A, B, C) \
  __builtin_amdgcn_wmma_f32_16x16x32_f16(false, (A), false, (B), (short)0, (C), false, false)

#if __has_builtin(__builtin_amdgcn_tensor_load_to_lds) && \
    __has_builtin(__builtin_amdgcn_s_wait_tensorcnt)
#define USE_TDM 1
#else
#define USE_TDM 0
#endif

#if USE_TDM
// Issue a TDM 2D tile load (f16 elements): tile0 contiguous cols, tile1 rows,
// row stride = stride0 elements. D# layout per CDNA5 ISA.
// amdgpu-toolchain (clang-23) arity: (v4u, v8i, v4i, v4i, v8i, i32 cpol).
__device__ __forceinline__ void tdm_load_tile_2d(
    const void* gsrc, unsigned ldsOff, int dim0, int dim1,
    int tile0, int tile1, int stride0) {
  unsigned long long ga = (unsigned long long)gsrc;
  v4u g0;
  g0[0] = 1u;                                              // count=1, user mode
  g0[1] = ldsOff;                                          // lds_addr
  g0[2] = (unsigned)ga;                                    // global_addr[31:0]
  g0[3] = (unsigned)((ga >> 32) & 0x1FFFFFFull) | (2u << 30); // addr[56:32]|type=2
  v8i g1;
  g1[0] = 0x00010000;                                      // data_size=1 -> 2 bytes
  g1[1] = (int)(((unsigned)dim0 & 0xFFFFu) << 16);         // tensor_dim0[15:0]
  g1[2] = (int)((((unsigned)dim0 >> 16) & 0xFFFFu) |
                (((unsigned)dim1 & 0xFFFFu) << 16));       // dim0[31:16]|dim1[15:0]
  g1[3] = (int)((((unsigned)dim1 >> 16) & 0xFFFFu) |
                ((unsigned)tile0 << 16));                  // dim1[31:16]|tile_dim0
  g1[4] = tile1;                                           // tile_dim1 (tile_dim2=0)
  g1[5] = stride0;                                         // tensor_dim0_stride[31:0]
  g1[6] = 0;
  g1[7] = 0;
  v4i z4 = {};
  v8i z8 = {};
  __builtin_amdgcn_tensor_load_to_lds(g0, g1, z4, z4, z8, 0);
}
#endif

// ------------------------------ helpers ------------------------------------

__global__ __launch_bounds__(256) void f32_to_f16_kernel(
    const float* __restrict__ src, _Float16* __restrict__ dst, int n) {
  int i = blockIdx.x * 256 + threadIdx.x;
  if (i < n) dst[i] = (_Float16)src[i];
}

// LayerNorm over last dim C=256; one block (256 threads) per row; f16 output.
__global__ __launch_bounds__(256) void ln_kernel(
    const float* __restrict__ x, const float* __restrict__ g,
    const float* __restrict__ bta, _Float16* __restrict__ out16, float eps) {
  __shared__ float red[8];
  const int row = blockIdx.x;
  const int tid = threadIdx.x;
  const float v = x[(size_t)row * 256 + tid];

  float s = v;
  #pragma unroll
  for (int o = 16; o; o >>= 1) s += __shfl_xor(s, o, 32);
  if ((tid & 31) == 0) red[tid >> 5] = s;
  __syncthreads();
  float mean = 0.f;
  #pragma unroll
  for (int i = 0; i < 8; ++i) mean += red[i];
  mean *= (1.0f / 256.0f);
  __syncthreads();

  const float d = v - mean;
  float s2 = d * d;
  #pragma unroll
  for (int o = 16; o; o >>= 1) s2 += __shfl_xor(s2, o, 32);
  if ((tid & 31) == 0) red[tid >> 5] = s2;
  __syncthreads();
  float var = 0.f;
  #pragma unroll
  for (int i = 0; i < 8; ++i) var += red[i];
  var *= (1.0f / 256.0f);

  const float y = d * rsqrtf(var + eps) * g[tid] + bta[tid];
  out16[(size_t)row * 256 + tid] = (_Float16)y;
}

// Gather 2x2 patches of xn1 into the (B*NR, 4*C) SR-conv A-matrix.
__global__ __launch_bounds__(256) void patch_kernel(
    const _Float16* __restrict__ xn1, _Float16* __restrict__ patches) {
  const size_t idx = (size_t)blockIdx.x * 256 + threadIdx.x; // 4096*1024
  const int kk = (int)(idx & 1023);
  const int row = (int)(idx >> 10);
  const int ci = kk & 255;
  const int pj = kk >> 8;            // i*2+j
  const int i = pj >> 1, j = pj & 1;
  const int b = row >> 10;
  const int rr = row & 1023;
  const int hr = rr >> 5, wr = rr & 31;
  patches[idx] =
      xn1[(((size_t)b * 64 + (2 * hr + i)) * 64 + (2 * wr + j)) * 256 + ci];
}

// ------------------------------ WMMA GEMM ----------------------------------
// C(MxN) = A(MxK)@W(KxN) + bias [+ resid], A/W row-major f16.
// 128 threads = 4 waves; each wave owns a 16x64 output tile. The 32x64 weight
// slab for each k-step is staged into LDS by the TDM (double-buffered,
// pipelined on TENSORcnt); all 4 waves read B-fragments from LDS.
template <bool OUT16, bool RES>
__global__ __launch_bounds__(128) void gemm_wmma_kernel(
    const _Float16* __restrict__ A, const _Float16* __restrict__ W,
    const float* __restrict__ bias, _Float16* __restrict__ out16,
    float* __restrict__ out32, const float* __restrict__ resid,
    int M, int N, int K) {
  __shared__ _Float16 wbuf[2][32][64];
  const int lane = threadIdx.x & 31;
  const int wave = threadIdx.x >> 5;
  const int hi   = lane >> 4;      // half-wave select
  const int l15  = lane & 15;
  const int mBase = blockIdx.y * 64 + wave * 16;
  const int nBase = blockIdx.x * 64;
  const int nk = K >> 5;

  v8f acc[4] = {};
  const int mRow = mBase + l15;    // A-frag: lane holds row (lane&15)
  const int sA   = hi * 16;

#if USE_TDM
  const unsigned lds0 = (unsigned)(unsigned long long)&wbuf[0][0][0];
  const unsigned lds1 = (unsigned)(unsigned long long)&wbuf[1][0][0];
  if (wave == 0) {  // prologue: slab 0
    tdm_load_tile_2d(W + nBase, lds0, N, K, 64, 32, N);
  }
#endif

  for (int it = 0; it < nk; ++it) {
    const int buf = it & 1;
    const int kk = it * 32;

#if USE_TDM
    if (wave == 0) {
      if (it + 1 < nk) {  // prefetch next slab into the other buffer
        tdm_load_tile_2d(W + (size_t)(kk + 32) * N + nBase,
                         (it & 1) ? lds0 : lds1, N, K, 64, 32, N);
        __builtin_amdgcn_s_wait_tensorcnt(1);   // slab `it` has landed
      } else {
        __builtin_amdgcn_s_wait_tensorcnt(0);
      }
    }
#else
    {   // cooperative fallback copy: 128 threads x 16 contiguous halves
      const _Float16* src = W + (size_t)kk * N + nBase;
      const int r = threadIdx.x >> 2;
      const int c0 = (threadIdx.x & 3) * 16;
      #pragma unroll
      for (int u = 0; u < 16; ++u) wbuf[buf][r][c0 + u] = src[(size_t)r * N + c0 + u];
    }
#endif
    __syncthreads();

    v16h a;
    const _Float16* ap = A + (size_t)mRow * K + kk + sA;
    __builtin_prefetch(ap + 32, 0, 1);
    #pragma unroll
    for (int w = 0; w < 8; ++w) { a[w] = ap[w]; a[8 + w] = ap[8 + w]; }

    #pragma unroll
    for (int t = 0; t < 4; ++t) {
      v16h bf;                     // B-frag: lane holds col l15, K = hi*16 + p
      #pragma unroll
      for (int p = 0; p < 16; ++p) bf[p] = wbuf[buf][hi * 16 + p][t * 16 + l15];
      acc[t] = WMMA_F16(a, bf, acc[t]);
    }
    __syncthreads();
  }

  #pragma unroll
  for (int t = 0; t < 4; ++t) {
    const int n = nBase + t * 16 + l15;
    const float bn = bias[n];
    #pragma unroll
    for (int r = 0; r < 8; ++r) {  // C/D: VGPR r = row r+8*hi, col = lane&15
      const int m = mBase + r + hi * 8;
      float v = acc[t][r] + bn;
      if (RES) v += resid[(size_t)m * N + n];
      if (OUT16) out16[(size_t)m * N + n] = (_Float16)v;
      else       out32[(size_t)m * N + n] = v;
    }
  }
}

// --------------------------- fused attention -------------------------------
// One wave per (b, head, 16-row q block); flash-attention over NR=1024 kv in
// 64-wide tiles (4 score WMMAs + 4 output WMMAs per tile, one softmax pass).
__global__ __launch_bounds__(128) void attn_kernel(
    const _Float16* __restrict__ q16, const _Float16* __restrict__ k16,
    const _Float16* __restrict__ v16, _Float16* __restrict__ o16) {
  __shared__ _Float16 lds[4][16][72];   // per-wave P tile, padded stride
  const int lane = threadIdx.x & 31;
  const int wave = threadIdx.x >> 5;
  const int gw   = blockIdx.x * 4 + wave;   // 0..8191
  const int b    = gw >> 11;
  const int h    = (gw >> 8) & 7;
  const int qblk = gw & 255;
  const int hi   = lane >> 4;
  const int l15  = lane & 15;
  const int sA   = hi * 16;
  const int qoff = qblk * 16;
  const float scale = 0.17677669529663687f;   // 32^-0.5

  // q A-fragment (16 rows x 32 head dims), loaded once.
  v16h aQ;
  {
    const _Float16* qp =
        q16 + ((size_t)(b * 4096 + qoff + l15)) * 256 + h * 32 + sA;
    #pragma unroll
    for (int w = 0; w < 8; ++w) { aQ[w] = qp[w]; aQ[8 + w] = qp[8 + w]; }
  }

  v8f accO0 = {}, accO1 = {};
  float mrow[8], lrow[8];
  #pragma unroll
  for (int r = 0; r < 8; ++r) { mrow[r] = -1e30f; lrow[r] = 0.0f; }

  for (int kv = 0; kv < 1024; kv += 64) {
    // Scores: 4 x (q(16x32) x K^T(32x16)).  B[d][n] = K[kv+c*16+n][h*32+d].
    v8f s[4];
    #pragma unroll
    for (int c = 0; c < 4; ++c) {
      const _Float16* kp =
          k16 + ((size_t)(b * 1024 + kv + c * 16 + l15)) * 256 + h * 32 + hi * 16;
      v16h bK;
      #pragma unroll
      for (int p = 0; p < 16; ++p) bK[p] = kp[p];
      v8f z = {};
      s[c] = WMMA_F16(aQ, bK, z);
    }

    // Online softmax; each output row lives in one VGPR across 16 lanes.
    float p[4][8];
    #pragma unroll
    for (int r = 0; r < 8; ++r) {
      const float a0 = s[0][r] * scale, a1 = s[1][r] * scale;
      const float a2 = s[2][r] * scale, a3 = s[3][r] * scale;
      float t = fmaxf(fmaxf(a0, a1), fmaxf(a2, a3));
      t = fmaxf(t, __shfl_xor(t, 1, 32));
      t = fmaxf(t, __shfl_xor(t, 2, 32));
      t = fmaxf(t, __shfl_xor(t, 4, 32));
      t = fmaxf(t, __shfl_xor(t, 8, 32));
      const float mn = fmaxf(mrow[r], t);
      const float alpha = __expf(mrow[r] - mn);
      mrow[r] = mn;
      p[0][r] = __expf(a0 - mn);
      p[1][r] = __expf(a1 - mn);
      p[2][r] = __expf(a2 - mn);
      p[3][r] = __expf(a3 - mn);
      float ts = (p[0][r] + p[1][r]) + (p[2][r] + p[3][r]);
      ts += __shfl_xor(ts, 1, 32);
      ts += __shfl_xor(ts, 2, 32);
      ts += __shfl_xor(ts, 4, 32);
      ts += __shfl_xor(ts, 8, 32);
      lrow[r] = lrow[r] * alpha + ts;
      accO0[r] *= alpha;
      accO1[r] *= alpha;
    }

    // Re-layout P (C/D layout -> A layout) through LDS.
    __syncthreads();
    #pragma unroll
    for (int r = 0; r < 8; ++r) {
      #pragma unroll
      for (int c = 0; c < 4; ++c)
        lds[wave][r + hi * 8][c * 16 + l15] = (_Float16)p[c][r];
    }
    __syncthreads();

    // Output: P(16x64) x V(64x32) as two 32-deep WMMA pairs.
    #pragma unroll
    for (int half = 0; half < 2; ++half) {
      const int base = half * 32;
      v16h aP;
      #pragma unroll
      for (int w = 0; w < 8; ++w) {
        aP[w]     = lds[wave][l15][base + sA + w];
        aP[8 + w] = lds[wave][l15][base + sA + 8 + w];
      }
      const _Float16* vp =
          v16 + ((size_t)(b * 1024 + kv + base + hi * 16)) * 256 + h * 32;
      v16h bV0, bV1;
      #pragma unroll
      for (int p2 = 0; p2 < 16; ++p2) {
        bV0[p2] = vp[(size_t)p2 * 256 + l15];
        bV1[p2] = vp[(size_t)p2 * 256 + 16 + l15];
      }
      accO0 = WMMA_F16(aP, bV0, accO0);
      accO1 = WMMA_F16(aP, bV1, accO1);
    }
  }

  #pragma unroll
  for (int r = 0; r < 8; ++r) {
    const float inv = 1.0f / lrow[r];
    const int m = qoff + r + hi * 8;
    _Float16* op = o16 + ((size_t)(b * 4096 + m)) * 256 + h * 32;
    op[l15]      = (_Float16)(accO0[r] * inv);
    op[16 + l15] = (_Float16)(accO1[r] * inv);
  }
}

// -------------------- depthwise 3x3 SAME + GELU(tanh) ----------------------
__global__ __launch_bounds__(256) void dwconv_gelu_kernel(
    const _Float16* __restrict__ hin, const float* __restrict__ wdw,
    const float* __restrict__ bdw, _Float16* __restrict__ hout) {
  const size_t idx = (size_t)blockIdx.x * 256 + threadIdx.x; // B*N*MID
  const int c = (int)(idx & 1023);
  const size_t pix = idx >> 10;
  const int xq = (int)(pix & 63);
  const int y  = (int)((pix >> 6) & 63);
  const int b  = (int)(pix >> 12);
  float acc = bdw[c];
  #pragma unroll
  for (int ky = 0; ky < 3; ++ky) {
    const int yy = y + ky - 1;
    if (yy < 0 || yy > 63) continue;
    #pragma unroll
    for (int kx = 0; kx < 3; ++kx) {
      const int xx = xq + kx - 1;
      if (xx < 0 || xx > 63) continue;
      acc += (float)hin[(((size_t)b * 64 + yy) * 64 + xx) * 1024 + c] *
             wdw[(ky * 3 + kx) * 1024 + c];
    }
  }
  const float u = acc;
  const float g =
      0.5f * u * (1.0f + tanhf(0.7978845608028654f * (u + 0.044715f * u * u * u)));
  hout[idx] = (_Float16)g;
}

// ------------------------------ launcher -----------------------------------

extern "C" void kernel_launch(void* const* d_in, const int* in_sizes, int n_in,
                              void* d_out, int out_size, void* d_ws, size_t ws_size,
                              hipStream_t stream) {
  const float* x    = (const float*)d_in[0];
  const float* g1   = (const float*)d_in[1];
  const float* b1   = (const float*)d_in[2];
  const float* g2   = (const float*)d_in[3];
  const float* b2   = (const float*)d_in[4];
  const float* wq   = (const float*)d_in[5];
  const float* bq   = (const float*)d_in[6];
  const float* wk   = (const float*)d_in[7];
  const float* bk   = (const float*)d_in[8];
  const float* wv   = (const float*)d_in[9];
  const float* bv   = (const float*)d_in[10];
  const float* wp   = (const float*)d_in[11];
  const float* bp   = (const float*)d_in[12];
  const float* wsr  = (const float*)d_in[13];
  const float* bsr  = (const float*)d_in[14];
  const float* gsr  = (const float*)d_in[15];
  const float* bsrn = (const float*)d_in[16];
  const float* wfc1 = (const float*)d_in[17];
  const float* bfc1 = (const float*)d_in[18];
  const float* wdw  = (const float*)d_in[19];
  const float* bdw  = (const float*)d_in[20];
  const float* wfc2 = (const float*)d_in[21];
  const float* bfc2 = (const float*)d_in[22];
  float* out = (float*)d_out;                 // also holds x1 (post-attn residual)

  const int BN = 4 * 4096;                    // 16384 full-seq rows
  const int BNR = 4 * 1024;                   // 4096 reduced rows

  char* wsp = (char*)d_ws;
  auto alloc = [&](size_t bytes) -> void* {
    void* p = (void*)wsp;
    wsp += (bytes + 255) & ~(size_t)255;
    return p;
  };
  _Float16* xn1    = (_Float16*)alloc((size_t)BN * 256 * 2);
  _Float16* q16    = (_Float16*)alloc((size_t)BN * 256 * 2);
  _Float16* patches= (_Float16*)alloc((size_t)BNR * 1024 * 2);
  float*    xr32   = (float*)   alloc((size_t)BNR * 256 * 4);
  _Float16* xrn    = (_Float16*)alloc((size_t)BNR * 256 * 2);
  _Float16* k16    = (_Float16*)alloc((size_t)BNR * 256 * 2);
  _Float16* v16    = (_Float16*)alloc((size_t)BNR * 256 * 2);
  _Float16* attn16 = (_Float16*)alloc((size_t)BN * 256 * 2);
  _Float16* xn2    = (_Float16*)alloc((size_t)BN * 256 * 2);
  _Float16* hmid   = (_Float16*)alloc((size_t)BN * 1024 * 2);
  _Float16* h2     = (_Float16*)alloc((size_t)BN * 1024 * 2);
  _Float16* wq16   = (_Float16*)alloc(65536 * 2);
  _Float16* wk16   = (_Float16*)alloc(65536 * 2);
  _Float16* wv16   = (_Float16*)alloc(65536 * 2);
  _Float16* wp16   = (_Float16*)alloc(65536 * 2);
  _Float16* wsr16  = (_Float16*)alloc(262144 * 2);
  _Float16* wfc1h  = (_Float16*)alloc(262144 * 2);
  _Float16* wfc2h  = (_Float16*)alloc(262144 * 2);

  // Weights -> f16
  f32_to_f16_kernel<<<256, 256, 0, stream>>>(wq, wq16, 65536);
  f32_to_f16_kernel<<<256, 256, 0, stream>>>(wk, wk16, 65536);
  f32_to_f16_kernel<<<256, 256, 0, stream>>>(wv, wv16, 65536);
  f32_to_f16_kernel<<<256, 256, 0, stream>>>(wp, wp16, 65536);
  f32_to_f16_kernel<<<1024, 256, 0, stream>>>(wsr, wsr16, 262144);
  f32_to_f16_kernel<<<1024, 256, 0, stream>>>(wfc1, wfc1h, 262144);
  f32_to_f16_kernel<<<1024, 256, 0, stream>>>(wfc2, wfc2h, 262144);

  // norm1
  ln_kernel<<<BN, 256, 0, stream>>>(x, g1, b1, xn1, 1e-6f);

  // q = xn1 @ wq + bq
  gemm_wmma_kernel<true, false><<<dim3(256 / 64, BN / 64), 128, 0, stream>>>(
      xn1, wq16, bq, q16, nullptr, nullptr, BN, 256, 256);

  // SR conv as GEMM over 2x2 patches, then LN(1e-5)
  patch_kernel<<<(BNR * 1024) / 256, 256, 0, stream>>>(xn1, patches);
  gemm_wmma_kernel<false, false><<<dim3(256 / 64, BNR / 64), 128, 0, stream>>>(
      patches, wsr16, bsr, nullptr, xr32, nullptr, BNR, 256, 1024);
  ln_kernel<<<BNR, 256, 0, stream>>>(xr32, gsr, bsrn, xrn, 1e-5f);

  // k, v
  gemm_wmma_kernel<true, false><<<dim3(256 / 64, BNR / 64), 128, 0, stream>>>(
      xrn, wk16, bk, k16, nullptr, nullptr, BNR, 256, 256);
  gemm_wmma_kernel<true, false><<<dim3(256 / 64, BNR / 64), 128, 0, stream>>>(
      xrn, wv16, bv, v16, nullptr, nullptr, BNR, 256, 256);

  // fused flash attention: 8192 waves, 4 waves/block
  attn_kernel<<<2048, 128, 0, stream>>>(q16, k16, v16, attn16);

  // out-proj + residual -> x1 (stored in d_out)
  gemm_wmma_kernel<false, true><<<dim3(256 / 64, BN / 64), 128, 0, stream>>>(
      attn16, wp16, bp, nullptr, out, x, BN, 256, 256);

  // norm2
  ln_kernel<<<BN, 256, 0, stream>>>(out, g2, b2, xn2, 1e-6f);

  // fc1 -> depthwise 3x3 + GELU -> fc2 + residual(x1) -> d_out
  gemm_wmma_kernel<true, false><<<dim3(1024 / 64, BN / 64), 128, 0, stream>>>(
      xn2, wfc1h, bfc1, hmid, nullptr, nullptr, BN, 1024, 256);
  dwconv_gelu_kernel<<<((size_t)BN * 1024) / 256, 256, 0, stream>>>(
      hmid, wdw, bdw, h2);
  gemm_wmma_kernel<false, true><<<dim3(256 / 64, BN / 64), 128, 0, stream>>>(
      h2, wfc2h, bfc2, nullptr, out, out, BN, 256, 1024);

  (void)in_sizes; (void)n_in; (void)out_size; (void)ws_size;
}
